// CompressionModule_27925877358821
// MI455X (gfx1250) — matrix operand order repaired and checked by reading, not compiled, not run on hardware
//
#include <hip/hip_runtime.h>

typedef __attribute__((ext_vector_type(16))) _Float16 v16h;
typedef __attribute__((ext_vector_type(8)))  _Float16 v8h;
typedef __attribute__((ext_vector_type(8)))  float    v8f;
typedef __attribute__((ext_vector_type(2)))  float    v2f;

#define WAVES_PER_BLOCK 8
#define THREADS (WAVES_PER_BLOCK * 32)
#define N_ENC 262144

#if defined(__has_builtin)
#if __has_builtin(__builtin_amdgcn_wmma_f16_16x16x32_f16)
#define HAVE_WMMA_F16_D 1
#endif
#endif
#ifndef HAVE_WMMA_F16_D
#define HAVE_WMMA_F16_D 0
#endif

// ---------------- LDS layout (in _Float16 units) ----------------
// sW1t [256][32]  @ 0        (W1 transposed: Wt[n][k], pitch 32)
// sW2t [64][256]  @ 8192     (pitch 256)
// sW3t [64][64]   @ 24576    (pitch 64)
// sW4t [16][64]   @ 28672    (pitch 64, N padded 3->16 with zeros)
// sBias   float[400] @ 29696 halves: b1[256] | b2[64] | b3[64] | b4 padded[16]
// sBiasH  f16[256]   @ 30496 halves: b1 as f16 (layer1 C-operand splats)
// sBiasH3 f16[64]    @ 30752 halves: b3 as f16 (layer3 C-operand splats)
// per-wave scratch @ 30816 + wave*4096 (each wave owns 32 points = 2 M-tiles):
//   feats 32x32 (1024) | xbuf 2x16x32 (1024) | act2 2x16x64 (2048)
//   act3 2x16x64 (2048) overlaps feats+xbuf (both dead by then)
#define BIAS_OFF    29696
#define BIASH_OFF   30496
#define BIASH3_OFF  30752
#define SCRATCH_OFF 30816
#define LDS_HALVES  (SCRATCH_OFF + WAVES_PER_BLOCK * 4096)

// Instant-NGP geometric resolutions: floor(16 * 64^(l/15)) (compile-time)
__device__ constexpr int RESA[16] = {16, 21, 27, 36, 48, 64, 84, 111,
                                     147, 194, 256, 337, 445, 588, 776, 1024};

__device__ __forceinline__ v8f wmma16(v16h a, v16h b, v8f c) {
  return __builtin_amdgcn_wmma_f32_16x16x32_f16(false, a, false, b, (short)0, c,
                                                false, false);
}
#if HAVE_WMMA_F16_D
__device__ __forceinline__ v8h wmma16h(v16h a, v16h b, v8h c) {
  return __builtin_amdgcn_wmma_f16_16x16x32_f16(false, a, false, b, (short)0, c,
                                                false, false);
}
#endif

__device__ __forceinline__ v8f splat8f(float v) {
  v8f r;
#pragma unroll
  for (int i = 0; i < 8; ++i) r[i] = v;
  return r;
}
__device__ __forceinline__ v8h splat8h(_Float16 v) {
  v8h r;
#pragma unroll
  for (int i = 0; i < 8; ++i) r[i] = v;
  return r;
}

// A-fragment (16xK row-major f16 LDS buffer, pitch halves), K-chunk of 32 at kbase.
// ISA 7.12.2 16-bit A layout: lanes 0-15 hold K[0..7],K[16..23];
// lanes 16-31 hold K[8..15],K[24..31] for row M = lane&15.
__device__ __forceinline__ v16h load_a_frag(const _Float16* buf, int pitch,
                                            int kbase, int lane) {
  const int row = lane & 15;
  const int cb  = kbase + ((lane >> 4) << 3);
  const _Float16* p = buf + row * pitch;
  union { v16h v; v8h h[2]; } u;
  u.h[0] = *(const v8h*)(p + cb);        // ds_load_b128
  u.h[1] = *(const v8h*)(p + cb + 16);   // ds_load_b128
  return u.v;
}

// B-fragment from N-major f16 weights Wt[n][k] (pitch halves):
// lane (N = ntile*16 + lane&15) needs K[kbase+(lane>>4)*8 +0..7] and K[kbase+16+(lane>>4)*8 +0..7].
__device__ __forceinline__ v16h load_b_frag(const _Float16* wt, int pitch,
                                            int ntile, int kbase, int lane) {
  const int n = (ntile << 4) + (lane & 15);
  const int k = kbase + ((lane >> 4) << 3);
  const _Float16* p = wt + n * pitch + k;
  union { v16h v; v8h h[2]; } u;
  u.h[0] = *(const v8h*)(p);             // ds_load_b128
  u.h[1] = *(const v8h*)(p + 16);
  return u.v;
}

// Store f32 D tile (lane VGPR r -> M = r + 8*(lane>>4), N = lane&15) with relu.
// fmaxf stays compiler-visible so the backend inserts WMMA->VALU hazard slots.
__device__ __forceinline__ void store_tile_relu(_Float16* buf, int pitch,
                                                int colbase, v8f c, int lane) {
  const int n = colbase + (lane & 15);
  const int rb = (lane >> 4) << 3;
#pragma unroll
  for (int r = 0; r < 8; ++r) {
    float v = fmaxf(c[r], 0.f);
    buf[(rb + r) * pitch + n] = (_Float16)v;
  }
}

#if HAVE_WMMA_F16_D
// Store f16 D tile (16-bit C/D layout: half e of lane -> M = e + 8*(lane>>4),
// N = lane&15) with packed relu; bias pre-added via C operand. Uses
// __builtin_fmaxf16 (compiler-visible) so the backend handles the WMMA->VALU
// RAW hazard itself; the extra canonicalize pk_max is the price of safety.
__device__ __forceinline__ void store_tile_relu_h(_Float16* buf, int pitch,
                                                  int colbase, v8h d, int lane) {
  const int n = colbase + (lane & 15);
  const int rb = (lane >> 4) << 3;
#pragma unroll
  for (int e = 0; e < 8; ++e) {
    _Float16 v = __builtin_fmaxf16(d[e], (_Float16)0.f);  // v_pk_max_num_f16
    buf[(rb + e) * pitch + n] = v;
  }
}
#endif

__global__ __launch_bounds__(THREADS) void ngp_fused(
    const float* __restrict__ xn, const float* __restrict__ tables,
    const float* __restrict__ W1, const float* __restrict__ b1,
    const float* __restrict__ W2, const float* __restrict__ b2,
    const float* __restrict__ W3, const float* __restrict__ b3,
    const float* __restrict__ W4, const float* __restrict__ b4,
    float* __restrict__ out, int nTiles) {
  extern __shared__ _Float16 lds[];
  _Float16* sW1 = lds;                 // 8192
  _Float16* sW2 = lds + 8192;          // 16384
  _Float16* sW3 = lds + 24576;         // 4096
  _Float16* sW4 = lds + 28672;         // 1024
  float*    sBias   = (float*)(lds + BIAS_OFF);  // 400 f32
  _Float16* sBiasH  = lds + BIASH_OFF;           // 256 f16 (b1)
  _Float16* sBiasH3 = lds + BIASH3_OFF;          // 64 f16 (b3)

  const int tid = threadIdx.x;

  // ---- cooperative f32 -> f16 weight transpose (+ biases) into LDS ----
  for (int t = tid; t < 8192; t += THREADS) {
    int n = t >> 5, k = t & 31;
    sW1[t] = (_Float16)W1[k * 256 + n];
  }
  for (int t = tid; t < 16384; t += THREADS) {
    int n = t >> 8, k = t & 255;
    sW2[t] = (_Float16)W2[k * 64 + n];
  }
  for (int t = tid; t < 4096; t += THREADS) {
    int n = t >> 6, k = t & 63;
    sW3[t] = (_Float16)W3[k * 64 + n];
  }
  for (int t = tid; t < 1024; t += THREADS) {
    int n = t >> 6, k = t & 63;
    sW4[t] = (n < 3) ? (_Float16)W4[k * 3 + n] : (_Float16)0.f;
  }
  for (int t = tid; t < 400; t += THREADS) {
    float v;
    if (t < 256)       v = b1[t];
    else if (t < 320)  v = b2[t - 256];
    else if (t < 384)  v = b3[t - 320];
    else               v = (t - 384 < 3) ? b4[t - 384] : 0.f;
    sBias[t] = v;
  }
  for (int t = tid; t < 256; t += THREADS) sBiasH[t] = (_Float16)b1[t];
  for (int t = tid; t < 64; t += THREADS) sBiasH3[t] = (_Float16)b3[t];
  __syncthreads();

  const int lane = tid & 31;
  const int wave = tid >> 5;
  _Float16* scratch = lds + SCRATCH_OFF + wave * 4096;
  _Float16* feats = scratch;          // 32x32 (two 16x32 A tiles)
  _Float16* xbuf  = scratch + 1024;   // two 16x32 chunks
  _Float16* act2  = scratch + 2048;   // two 16x64
  _Float16* act3  = scratch;          // two 16x64, reuses feats+xbuf

  const int wavesTotal = (int)gridDim.x * WAVES_PER_BLOCK;
  const int waveId = (int)blockIdx.x * WAVES_PER_BLOCK + wave;
  const int n1 = lane & 15;  // column within a 16-wide N tile (per-lane)

  for (int tile = waveId; tile < nTiles; tile += wavesTotal) {
    const int pbase = tile << 5;  // 32 points per wave-tile

    {
      const int nxt = tile + wavesTotal;
      if (nxt < nTiles)
        __builtin_prefetch(xn + (size_t)(nxt << 5) * 2, 0, 1);  // global_prefetch_b8
    }

    // ---- hash encoding: one point per lane, levels fully unrolled ----
    {
      const v2f xy = *(const v2f*)(xn + (size_t)(pbase + lane) * 2);
      const float x01 = (xy.x + 1.f) * 0.5f;
      const float y01 = (xy.y + 1.f) * 0.5f;
      v8h fp[4];  // 32 packed f16 features (levels 0..15 x 2)
#pragma unroll
      for (int lvl = 0; lvl < 16; ++lvl) {
        const int res = RESA[lvl];            // immediate after unroll
        const float rf = (float)res;
        const float sx = x01 * (rf - 1.f);
        const float sy = y01 * (rf - 1.f);
        float fx0 = fminf(fmaxf(floorf(sx), 0.f), rf - 2.f);
        float fy0 = fminf(fmaxf(floorf(sy), 0.f), rf - 2.f);
        const int ix0 = (int)fx0, iy0 = (int)fy0;
        const float fx = sx - fx0, fy = sy - fy0;
        const bool dense = (res * res) <= N_ENC;  // compile-time
        const float* tab = tables + (size_t)lvl * (2u * N_ENC);
        float f0 = 0.f, f1 = 0.f;
#pragma unroll
        for (int c = 0; c < 4; ++c) {
          const int dx = c & 1, dy = c >> 1;
          const int ix = ix0 + dx, iy = iy0 + dy;
          int gidx;
          if (dense) {
            gidx = iy * res + ix;
          } else {
            gidx = (int)(((unsigned)ix ^ ((unsigned)iy * 2654435761u)) &
                         (N_ENC - 1u));
          }
          const v2f ft = *(const v2f*)(tab + (size_t)gidx * 2);  // L2 gather
          const float w = (dx ? fx : 1.f - fx) * (dy ? fy : 1.f - fy);
          f0 += ft.x * w;
          f1 += ft.y * w;
        }
        fp[lvl >> 2][(lvl & 3) * 2]     = (_Float16)f0;
        fp[lvl >> 2][(lvl & 3) * 2 + 1] = (_Float16)f1;
      }
      _Float16* frow = feats + lane * 32;
#pragma unroll
      for (int i = 0; i < 4; ++i)
        *(v8h*)(frow + i * 8) = fp[i];      // 4x ds_store_b128
    }

    const v16h aF0 = load_a_frag(feats, 32, 0, lane);        // points 0-15
    const v16h aF1 = load_a_frag(feats + 512, 32, 0, lane);  // points 16-31

    // ---- layer1 (32->256) fused with layer2 (256->64) K-accumulation ----
    v8f acc2a[4], acc2b[4];
#pragma unroll
    for (int nt = 0; nt < 4; ++nt) {
      const v8f binit = splat8f(sBias[256 + nt * 16 + n1]);
      acc2a[nt] = binit;
      acc2b[nt] = binit;
    }
#pragma unroll 1
    for (int kc = 0; kc < 8; ++kc) {
#pragma unroll
      for (int h = 0; h < 2; ++h) {
        const int nt1 = kc * 2 + h;
        const v16h bF = load_b_frag(sW1, 32, nt1, 0, lane);
#if HAVE_WMMA_F16_D
        const v8h cb = splat8h(sBiasH[nt1 * 16 + n1]);  // bias via C operand
        v8h d0 = wmma16h(aF0, bF, cb);
        v8h d1 = wmma16h(aF1, bF, cb);
        store_tile_relu_h(xbuf, 32, h * 16, d0, lane);
        store_tile_relu_h(xbuf + 512, 32, h * 16, d1, lane);
#else
        const v8f cb = splat8f(sBias[nt1 * 16 + n1]);
        v8f c0 = wmma16(aF0, bF, cb);
        v8f c1 = wmma16(aF1, bF, cb);
        store_tile_relu(xbuf, 32, h * 16, c0, lane);
        store_tile_relu(xbuf + 512, 32, h * 16, c1, lane);
#endif
      }
      const v16h a2x = load_a_frag(xbuf, 32, 0, lane);
      const v16h a2y = load_a_frag(xbuf + 512, 32, 0, lane);
#pragma unroll
      for (int nt = 0; nt < 4; ++nt) {
        const v16h bF = load_b_frag(sW2, 256, nt, kc * 32, lane);
        acc2a[nt] = wmma16(a2x, bF, acc2a[nt]);
        acc2b[nt] = wmma16(a2y, bF, acc2b[nt]);
      }
    }
#pragma unroll
    for (int nt = 0; nt < 4; ++nt) {
      store_tile_relu(act2, 64, nt * 16, acc2a[nt], lane);
      store_tile_relu(act2 + 1024, 64, nt * 16, acc2b[nt], lane);
    }

    // ---- layer3 (64->64): f16 accumulation (2-WMMA chain) ----
#if HAVE_WMMA_F16_D
    v8h acc3a[4], acc3b[4];
#pragma unroll
    for (int nt = 0; nt < 4; ++nt) {
      const v8h binit = splat8h(sBiasH3[nt * 16 + n1]);
      acc3a[nt] = binit;
      acc3b[nt] = binit;
    }
#pragma unroll
    for (int kc = 0; kc < 2; ++kc) {
      const v16h a3x = load_a_frag(act2, 64, kc * 32, lane);
      const v16h a3y = load_a_frag(act2 + 1024, 64, kc * 32, lane);
#pragma unroll
      for (int nt = 0; nt < 4; ++nt) {
        const v16h bF = load_b_frag(sW3, 64, nt, kc * 32, lane);
        acc3a[nt] = wmma16h(a3x, bF, acc3a[nt]);
        acc3b[nt] = wmma16h(a3y, bF, acc3b[nt]);
      }
    }
#pragma unroll
    for (int nt = 0; nt < 4; ++nt) {
      store_tile_relu_h(act3, 64, nt * 16, acc3a[nt], lane);
      store_tile_relu_h(act3 + 1024, 64, nt * 16, acc3b[nt], lane);
    }
#else
    v8f acc3a[4], acc3b[4];
#pragma unroll
    for (int nt = 0; nt < 4; ++nt) {
      const v8f binit = splat8f(sBias[320 + nt * 16 + n1]);
      acc3a[nt] = binit;
      acc3b[nt] = binit;
    }
#pragma unroll
    for (int kc = 0; kc < 2; ++kc) {
      const v16h a3x = load_a_frag(act2, 64, kc * 32, lane);
      const v16h a3y = load_a_frag(act2 + 1024, 64, kc * 32, lane);
#pragma unroll
      for (int nt = 0; nt < 4; ++nt) {
        const v16h bF = load_b_frag(sW3, 64, nt, kc * 32, lane);
        acc3a[nt] = wmma16(a3x, bF, acc3a[nt]);
        acc3b[nt] = wmma16(a3y, bF, acc3b[nt]);
      }
    }
#pragma unroll
    for (int nt = 0; nt < 4; ++nt) {
      store_tile_relu(act3, 64, nt * 16, acc3a[nt], lane);
      store_tile_relu(act3 + 1024, 64, nt * 16, acc3b[nt], lane);
    }
#endif

    // ---- layer4 (64->3, N padded to 16): f32 output ----
    v8f acc4a = splat8f(sBias[384 + n1]);
    v8f acc4b = acc4a;
#pragma unroll
    for (int kc = 0; kc < 2; ++kc) {
      const v16h a4x = load_a_frag(act3, 64, kc * 32, lane);
      const v16h a4y = load_a_frag(act3 + 1024, 64, kc * 32, lane);
      const v16h bF = load_b_frag(sW4, 64, 0, kc * 32, lane);
      acc4a = wmma16(a4x, bF, acc4a);
      acc4b = wmma16(a4y, bF, acc4b);
    }
    if (n1 < 3) {
      const int rb = (lane >> 4) << 3;
#pragma unroll
      for (int r = 0; r < 8; ++r) {
        out[(size_t)(pbase + rb + r) * 3 + n1]      = acc4a[r];
        out[(size_t)(pbase + 16 + rb + r) * 3 + n1] = acc4b[r];
      }
    }
  }
}

extern "C" void kernel_launch(void* const* d_in, const int* in_sizes, int n_in,
                              void* d_out, int out_size, void* d_ws,
                              size_t ws_size, hipStream_t stream) {
  (void)n_in; (void)out_size; (void)d_ws; (void)ws_size;
  const float* xn     = (const float*)d_in[0];
  const float* tables = (const float*)d_in[1];
  const float* W1 = (const float*)d_in[2];
  const float* b1 = (const float*)d_in[3];
  const float* W2 = (const float*)d_in[4];
  const float* b2 = (const float*)d_in[5];
  const float* W3 = (const float*)d_in[6];
  const float* b3 = (const float*)d_in[7];
  const float* W4 = (const float*)d_in[8];
  const float* b4 = (const float*)d_in[9];
  float* out = (float*)d_out;

  const int B = in_sizes[0] / 2;
  const int nTiles = B / 32;  // B = 1048576 -> 32768 tiles of M=32

  const size_t shmem = (size_t)LDS_HALVES * sizeof(_Float16);  // ~124 KB
  (void)hipFuncSetAttribute((const void*)ngp_fused,
                            hipFuncAttributeMaxDynamicSharedMemorySize,
                            (int)shmem);

  int blocks = 2048;  // 16384 waves -> 2 tiles/wave
  if (blocks * WAVES_PER_BLOCK > nTiles)
    blocks = (nTiles + WAVES_PER_BLOCK - 1) / WAVES_PER_BLOCK;

  ngp_fused<<<blocks, THREADS, shmem, stream>>>(xn, tables, W1, b1, W2, b2, W3,
                                                b3, W4, b4, out, nTiles);
}
